// vanilaGCN_6004364280506
// MI455X (gfx1250) — matrix operand order repaired and checked by reading, compiled, not run on hardware
//
#include <hip/hip_runtime.h>
#include <hip/hip_bf16.h>

// ---------------------------------------------------------------------------
// WMMA fragment types
// ---------------------------------------------------------------------------
typedef __attribute__((ext_vector_type(16))) __bf16 v16bf;
typedef __attribute__((ext_vector_type(8)))  float  v8f;
typedef __attribute__((ext_vector_type(4)))  int    v4i;

union Frag16 {
    v16bf    bf;
    unsigned u[8];
};

#if __has_builtin(__builtin_amdgcn_global_load_async_to_lds_b128) && \
    __has_builtin(__builtin_amdgcn_s_wait_asynccnt)
#define USE_ASYNC_LDS 1
typedef __attribute__((address_space(1))) v4i gv4i;   // global int4
typedef __attribute__((address_space(3))) v4i lv4i;   // LDS int4
#endif

// round-to-nearest-even float -> bf16, two values packed into one dword
__device__ __forceinline__ unsigned pack_bf16(float lo, float hi) {
    unsigned a = __float_as_uint(lo);
    unsigned b = __float_as_uint(hi);
    a = (a + 0x7FFFu + ((a >> 16) & 1u)) >> 16;
    b = (b + 0x7FFFu + ((b >> 16) & 1u)) >> 16;
    return (b << 16) | (a & 0xFFFFu);
}

// ---------------------------------------------------------------------------
// f32 -> packed bf16 conversion pre-pass (8 elems / thread), optional ReLU.
// ---------------------------------------------------------------------------
__global__ __launch_bounds__(256)
void cvt_bf16x8(const float4* __restrict__ x, uint4* __restrict__ y,
                size_t n8, int applyRelu)
{
    size_t i = (size_t)blockIdx.x * blockDim.x + threadIdx.x;
    if (i >= n8) return;
    float4 a = x[2 * i];
    float4 b = x[2 * i + 1];
    if (applyRelu) {
        a.x = fmaxf(a.x, 0.f); a.y = fmaxf(a.y, 0.f);
        a.z = fmaxf(a.z, 0.f); a.w = fmaxf(a.w, 0.f);
        b.x = fmaxf(b.x, 0.f); b.y = fmaxf(b.y, 0.f);
        b.z = fmaxf(b.z, 0.f); b.w = fmaxf(b.w, 0.f);
    }
    uint4 o;
    o.x = pack_bf16(a.x, a.y);  o.y = pack_bf16(a.z, a.w);
    o.z = pack_bf16(b.x, b.y);  o.w = pack_bf16(b.z, b.w);
    y[i] = o;
}

// ---------------------------------------------------------------------------
// Pre-pack weights: Wp[kp*F + c] = {bf16(W[2kp+1][c]), bf16(W[2kp][c])}
// = exactly the B-fragment dword layout (VGPR j holds K = 2j, 2j+1).
// ---------------------------------------------------------------------------
__global__ __launch_bounds__(256)
void pack_w(const float* __restrict__ W, unsigned* __restrict__ Wp,
            int K2, int F)
{
    int i = blockIdx.x * blockDim.x + threadIdx.x;
    if (i >= K2 * F) return;
    int kp = i / F;
    int c  = i - kp * F;
    Wp[i] = pack_bf16(W[(size_t)(2 * kp) * F + c],
                      W[(size_t)(2 * kp + 1) * F + c]);
}

// ---------------------------------------------------------------------------
// WMMA GEMM:  H[M,F] = Xb[M,K] @ Wp[K,F]  (both already packed bf16)
// block = 256 threads (8 waves). Each wave: 16 rows x 64 cols (4 WMMA tiles).
// K-slab = 64 (8KB), double-buffered via GLOBAL_LOAD_ASYNC_TO_LDS_B128:
//   issue slab i+1 right after the barrier, compute slab i while it flies.
// One barrier per slab; ASYNCcnt completes in order so wait(0) at slab top
// guarantees slab i has landed.
// grid.x = ceil(M/128), grid.y = F/64.  K % 64 == 0, F % 64 == 0.
// ---------------------------------------------------------------------------
__global__ __launch_bounds__(256)
void gcn_gemm_wmma(const unsigned* __restrict__ Xb,  // [M][K/2] packed pairs
                   const unsigned* __restrict__ Wp,  // [K/2][F] packed pairs
                   float* __restrict__ H, int M, int K, int F)
{
    __shared__ unsigned lB[2][32 * 64];     // 2 x (32 K-pairs x 64 cols) = 16 KB

    const int tid   = threadIdx.x;
    const int wave  = tid >> 5;
    const int lane  = tid & 31;
    const int l15   = lane & 15;
    const int khalf = lane >> 4;            // 0 or 1

    const int blockM  = blockIdx.x * 128;
    const int colBase = blockIdx.y * 64;

    const int row  = blockM + wave * 16 + l15;
    const int rowc = row < M ? row : (M - 1);

    const int K2 = K >> 1;
    const unsigned* __restrict__ xr = Xb + (size_t)rowc * K2;

    // staging pointers: thread tid copies quad tid and quad tid+256 of a slab
    const size_t wstep = (size_t)32 * F;    // dwords advanced per slab
    const unsigned* w0 = Wp + (size_t)(tid >> 4) * F + colBase + ((tid & 15) << 2);
    const unsigned* w1 = w0 + (size_t)16 * F;

    v8f acc0 = {}; v8f acc1 = {}; v8f acc2 = {}; v8f acc3 = {};

    const int nslab = K >> 6;

    // ---- prologue: issue slab 0 into buffer 0 ----
#ifdef USE_ASYNC_LDS
    __builtin_amdgcn_global_load_async_to_lds_b128(
        (gv4i*)w0, (lv4i*)&lB[0][tid << 2], 0, 0);
    __builtin_amdgcn_global_load_async_to_lds_b128(
        (gv4i*)w1, (lv4i*)&lB[0][(tid + 256) << 2], 0, 0);
#else
    *(uint4*)&lB[0][tid << 2]         = *(const uint4*)w0;
    *(uint4*)&lB[0][(tid + 256) << 2] = *(const uint4*)w1;
#endif
    w0 += wstep;
    w1 += wstep;

    for (int i = 0; i < nslab; ++i) {
#ifdef USE_ASYNC_LDS
        __builtin_amdgcn_s_wait_asynccnt(0);    // slab i landed (in-order)
#endif
        __syncthreads();                        // visible to all waves

        // ---- issue slab i+1 into the other buffer (overlaps compute) ----
        if (i + 1 < nslab) {
            const int nb = (i + 1) & 1;
#ifdef USE_ASYNC_LDS
            __builtin_amdgcn_global_load_async_to_lds_b128(
                (gv4i*)w0, (lv4i*)&lB[nb][tid << 2], 0, 0);
            __builtin_amdgcn_global_load_async_to_lds_b128(
                (gv4i*)w1, (lv4i*)&lB[nb][(tid + 256) << 2], 0, 0);
#else
            *(uint4*)&lB[nb][tid << 2]         = *(const uint4*)w0;
            *(uint4*)&lB[nb][(tid + 256) << 2] = *(const uint4*)w1;
#endif
            w0 += wstep;
            w1 += wstep;
        }

        const unsigned* __restrict__ buf = lB[i & 1];
        const int kpSlab = i << 5;              // K-pair base of this slab

        #pragma unroll
        for (int s = 0; s < 2; ++s) {
            // ---- A fragment: 2x global_load_b128, k = 16*khalf + 0..15, same order as B ----
            Frag16 a;
            const uint4* xv =
                (const uint4*)(xr + kpSlab + s * 16 + khalf * 8);
            uint4 alo = xv[0];
            uint4 ahi = xv[1];                  // +4 dwords
            a.u[0] = alo.x; a.u[1] = alo.y; a.u[2] = alo.z; a.u[3] = alo.w;
            a.u[4] = ahi.x; a.u[5] = ahi.y; a.u[6] = ahi.z; a.u[7] = ahi.w;

            // speculative prefetch of the next A slab
            __builtin_prefetch(xr + kpSlab + 32, 0, 1);

            // ---- 4 B fragments + 4 WMMAs ----
            const int kbase = s * 16 + khalf * 8;
            Frag16 b0, b1, b2, b3;
            #pragma unroll
            for (int j = 0; j < 8; ++j) {
                const unsigned* r = &buf[(kbase + j) * 64 + l15];
                b0.u[j] = r[0];
                b1.u[j] = r[16];
                b2.u[j] = r[32];
                b3.u[j] = r[48];
            }
            acc0 = __builtin_amdgcn_wmma_f32_16x16x32_bf16(
                false, a.bf, false, b0.bf, (short)0, acc0, false, false);
            acc1 = __builtin_amdgcn_wmma_f32_16x16x32_bf16(
                false, a.bf, false, b1.bf, (short)0, acc1, false, false);
            acc2 = __builtin_amdgcn_wmma_f32_16x16x32_bf16(
                false, a.bf, false, b2.bf, (short)0, acc2, false, false);
            acc3 = __builtin_amdgcn_wmma_f32_16x16x32_bf16(
                false, a.bf, false, b3.bf, (short)0, acc3, false, false);
        }
    }

    // ---- store C (VGPR j -> row j (lanes 0-15) / j+8 (lanes 16-31)) ----
    const int rbase = blockM + wave * 16 + (lane < 16 ? 0 : 8);
    v8f accv[4] = { acc0, acc1, acc2, acc3 };
    if (blockM + 128 <= M) {                // fast path: whole tile in bounds
        float* __restrict__ hp = H + (size_t)rbase * F + colBase + l15;
        #pragma unroll
        for (int j = 0; j < 8; ++j) {
            float* __restrict__ hr = hp + (size_t)j * F;
            hr[0]  = accv[0][j];
            hr[16] = accv[1][j];
            hr[32] = accv[2][j];
            hr[48] = accv[3][j];
        }
    } else {
        #pragma unroll
        for (int j = 0; j < 8; ++j) {
            const int r = rbase + j;
            if (r < M) {
                float* __restrict__ hr = H + (size_t)r * F + colBase + l15;
                hr[0]  = accv[0][j];
                hr[16] = accv[1][j];
                hr[32] = accv[2][j];
                hr[48] = accv[3][j];
            }
        }
    }
}

// ---------------------------------------------------------------------------
// Degree / normalization kernels (run once; A is layer-invariant)
// ---------------------------------------------------------------------------
__global__ void deg_init(float* deg, int M) {
    int i = blockIdx.x * blockDim.x + threadIdx.x;
    if (i < M) deg[i] = 1.0f;                       // self-loop
}
__global__ void deg_accum(float* deg, const int* __restrict__ dst, int E) {
    int i = blockIdx.x * blockDim.x + threadIdx.x;
    if (i < E) atomicAdd(&deg[dst[i]], 1.0f);
}
__global__ void deg_rsqrt(float* deg, int M) {
    int i = blockIdx.x * blockDim.x + threadIdx.x;
    if (i < M) deg[i] = rsqrtf(deg[i]);             // in-place -> dinv
}

// ---------------------------------------------------------------------------
// agg[n][f] = h[n][f] * dinv[n]^2 + bias[f]   (float4 vectorized)
// ---------------------------------------------------------------------------
__global__ __launch_bounds__(256)
void agg_init(const float* __restrict__ h, const float* __restrict__ dinv,
              const float* __restrict__ bias, float* __restrict__ agg,
              size_t total4, int f4shift)
{
    size_t i = (size_t)blockIdx.x * blockDim.x + threadIdx.x;
    if (i >= total4) return;
    int n  = (int)(i >> f4shift);
    int f4 = (int)(i & (((size_t)1 << f4shift) - 1));
    float s = dinv[n]; s *= s;
    float4 hv = ((const float4*)h)[i];
    float4 bv = ((const float4*)bias)[f4];
    float4 o;
    o.x = hv.x * s + bv.x;  o.y = hv.y * s + bv.y;
    o.z = hv.z * s + bv.z;  o.w = hv.w * s + bv.w;
    ((float4*)agg)[i] = o;
}

// ---------------------------------------------------------------------------
// Edge scatter: one wave per edge; float4 gathers of h[src], f32 global
// atomics into agg[dst]. h fits in the 192MB L2.
// ---------------------------------------------------------------------------
__global__ __launch_bounds__(256)
void edge_scatter(const float* __restrict__ h, float* __restrict__ agg,
                  const int* __restrict__ src, const int* __restrict__ dst,
                  const float* __restrict__ dinv, int E, int F)
{
    int wid  = (int)(((size_t)blockIdx.x * blockDim.x + threadIdx.x) >> 5);
    int lane = threadIdx.x & 31;
    if (wid >= E) return;
    const int s = src[wid];
    const int d = dst[wid];
    const float norm = dinv[s] * dinv[d];
    const float4* __restrict__ hs = (const float4*)(h + (size_t)s * F);
    float* __restrict__ ad = agg + (size_t)d * F;
    const int nf4 = F >> 2;
    for (int f = lane; f < nf4; f += 32) {
        float4 v = hs[f];
        int b = f << 2;
        atomicAdd(ad + b + 0, v.x * norm);
        atomicAdd(ad + b + 1, v.y * norm);
        atomicAdd(ad + b + 2, v.z * norm);
        atomicAdd(ad + b + 3, v.w * norm);
    }
}

// ---------------------------------------------------------------------------
// log_softmax over 64 features, one wave per node (wave32 shuffle reduction)
// ---------------------------------------------------------------------------
__global__ __launch_bounds__(256)
void log_softmax64(const float* __restrict__ a, float* __restrict__ out, int M)
{
    int wid  = (int)(((size_t)blockIdx.x * blockDim.x + threadIdx.x) >> 5);
    int lane = threadIdx.x & 31;
    if (wid >= M) return;
    const float* r = a + (size_t)wid * 64;
    float v0 = r[lane];
    float v1 = r[lane + 32];
    float m = fmaxf(v0, v1);
    #pragma unroll
    for (int o = 16; o >= 1; o >>= 1) m = fmaxf(m, __shfl_xor(m, o, 32));
    float s = __expf(v0 - m) + __expf(v1 - m);
    #pragma unroll
    for (int o = 16; o >= 1; o >>= 1) s += __shfl_xor(s, o, 32);
    float ls = __logf(s) + m;
    float* w = out + (size_t)wid * 64;
    w[lane]      = v0 - ls;
    w[lane + 32] = v1 - ls;
}

// ---------------------------------------------------------------------------
// Orchestration
// ---------------------------------------------------------------------------
extern "C" void kernel_launch(void* const* d_in, const int* in_sizes, int n_in,
                              void* d_out, int out_size, void* d_ws, size_t ws_size,
                              hipStream_t stream)
{
    const float* x  = (const float*)d_in[0];
    const int*   ei = (const int*)  d_in[1];
    const float* W1 = (const float*)d_in[2];
    const float* b1 = (const float*)d_in[3];
    const float* W2 = (const float*)d_in[4];
    const float* b2 = (const float*)d_in[5];
    const float* W3 = (const float*)d_in[6];
    const float* b3 = (const float*)d_in[7];

    const int M = in_sizes[0] / 512;        // 100000 nodes
    const int E = in_sizes[1] / 2;          // 1.6M edges
    const int* srcI = ei;
    const int* dstI = ei + E;

    // workspace layout:
    //  [0]       dinv            (1 MB pad)
    //  [1MB]     hbuf  f32       (104 MB)
    //  [105MB]   abuf  f32       (104 MB)
    //  [209MB]   xb    bf16-pack (104 MB)
    //  [313MB]   wpack dwords    (256 KB)
    char* ws = (char*)d_ws;
    float*    dinv = (float*)ws;
    float*    hbuf = (float*)(ws + (1u << 20));
    float*    abuf = (float*)(ws + (1u << 20) + (size_t)104857600u);
    unsigned* xb   = (unsigned*)(ws + (1u << 20) + (size_t)104857600u * 2);
    unsigned* wp   = (unsigned*)(ws + (1u << 20) + (size_t)104857600u * 3);
    float*    outp = (float*)d_out;

    // --- degree / dinv (once) ---
    deg_init <<<(M + 255) / 256, 256, 0, stream>>>(dinv, M);
    deg_accum<<<(E + 255) / 256, 256, 0, stream>>>(dinv, dstI, E);
    deg_rsqrt<<<(M + 255) / 256, 256, 0, stream>>>(dinv, M);

    auto layer = [&](const float* act, int K, const float* Wt, const float* bt,
                     int F, int relu, int f4shift) {
        // activation f32 -> packed bf16 (ReLU fused for layers 2/3)
        size_t n8 = (size_t)M * K / 8;
        cvt_bf16x8<<<(unsigned)((n8 + 255) / 256), 256, 0, stream>>>(
            (const float4*)act, (uint4*)xb, n8, relu);
        // weight f32 -> packed bf16 pairs in B-fragment layout
        int k2f = (K / 2) * F;
        pack_w<<<(k2f + 255) / 256, 256, 0, stream>>>(Wt, wp, K / 2, F);
        // WMMA GEMM
        dim3 g((M + 127) / 128, F / 64);
        gcn_gemm_wmma<<<g, 256, 0, stream>>>(xb, wp, hbuf, M, K, F);
        // self-loop + bias init, then edge scatter
        size_t total4 = (size_t)M * F / 4;
        agg_init<<<(unsigned)((total4 + 255) / 256), 256, 0, stream>>>(
            hbuf, dinv, bt, abuf, total4, f4shift);
        edge_scatter<<<(E + 7) / 8, 256, 0, stream>>>(
            hbuf, abuf, srcI, dstI, dinv, E, F);
    };

    layer(x,    512, W1, b1, 256, 0, 6);
    layer(abuf, 256, W2, b2, 128, 1, 5);
    layer(abuf, 128, W3, b3,  64, 1, 4);

    log_softmax64<<<(M + 7) / 8, 256, 0, stream>>>(abuf, outp, M);
}
